// AbstractAttention_72086731096343
// MI455X (gfx1250) — matrix operand order repaired and checked by reading, not compiled, not run on hardware
//
#include <hip/hip_runtime.h>

// ---------------------------------------------------------------------------
// MHA (TransformerLens AbstractAttention) for gfx1250 / MI455X.
// All four matmuls on v_wmma_f32_16x16x32_bf16 (wave32), fp32 accumulate.
// GEMMs: 32(M)x64(N) per-wave tiles, ping-pong software-pipelined K loop
// (no buffer-rotation movs). Attention: flash-style, V fragments preloaded,
// base-2 softmax, next-block prefetch.
// ---------------------------------------------------------------------------

#define D_MODEL 1024
#define N_HEADS 16
#define D_HEAD  64
#define SEQ     2048
#define BATCH   2
#define NEG_INF (-1e9f)
// attention scale folded with log2(e): softmax(s/8) == 2^(s*0.125*log2e - m2)
#define SCALE_LOG2 (0.125f * 1.44269504088896f)

typedef __bf16 bf16_t;
typedef __attribute__((ext_vector_type(16))) __bf16 v16bf;
typedef __attribute__((ext_vector_type(8)))  __bf16 v8bf;
typedef __attribute__((ext_vector_type(8)))  float  v8f;

static __device__ inline v8f vzero8() {
  v8f z;
#pragma unroll
  for (int i = 0; i < 8; ++i) z[i] = 0.0f;
  return z;
}

static __device__ inline v8f wmma_bf16(v16bf a, v16bf b, v8f c) {
  // D(16x16,f32) = A(16x32,bf16) * B(32x16,bf16) + C
  return __builtin_amdgcn_wmma_f32_16x16x32_bf16(
      false, a, false, b, (short)0, c, false, false);
}

static __device__ inline v16bf pack16(v8bf lo, v8bf hi) {
  v16bf r;
#pragma unroll
  for (int i = 0; i < 8; ++i) { r[i] = lo[i]; r[8 + i] = hi[i]; }
  return r;
}

// A fragment (16x32 bf16). rowp = &A[m0 + (lane&15)][0] (row-major, any ld).
// lane<16: k kb+0..15; lane>=16: k kb+16..31 (contiguous).
static __device__ inline v16bf load_a_bf16(const bf16_t* rowp, int kb, int lane) {
  const int hi = (lane >> 4) << 4;                    // 0 or 16
  v8bf c0 = *(const v8bf*)(rowp + kb + hi);
  v8bf c1 = *(const v8bf*)(rowp + kb + hi + 8);
  return pack16(c0, c1);
}

// Same A fragment but converting fp32 -> bf16 on the fly.
static __device__ inline v16bf load_a_f32(const float* rowp, int kb, int lane) {
  const int hi = (lane >> 4) << 4;
  float4 x0 = *(const float4*)(rowp + kb + hi);
  float4 x1 = *(const float4*)(rowp + kb + hi + 4);
  float4 y0 = *(const float4*)(rowp + kb + hi + 8);
  float4 y1 = *(const float4*)(rowp + kb + hi + 12);
  v16bf a;
  a[0]=(__bf16)x0.x; a[1]=(__bf16)x0.y; a[2] =(__bf16)x0.z; a[3] =(__bf16)x0.w;
  a[4]=(__bf16)x1.x; a[5]=(__bf16)x1.y; a[6] =(__bf16)x1.z; a[7] =(__bf16)x1.w;
  a[8]=(__bf16)y0.x; a[9]=(__bf16)y0.y; a[10]=(__bf16)y0.z; a[11]=(__bf16)y0.w;
  a[12]=(__bf16)y1.x;a[13]=(__bf16)y1.y;a[14]=(__bf16)y1.z; a[15]=(__bf16)y1.w;
  return a;
}

// B fragment (32x16 bf16): lanes 0-15 hold K 0..15 of column (lane&15),
// lanes 16-31 hold K 16..31.  p = &B[ncol + (lane&15)][kb + ((lane>>4)<<4)]
// where B is stored N-major with K contiguous.  16 contiguous elements.
static __device__ inline v16bf load_b_frag(const bf16_t* p) {
  v8bf c0 = *(const v8bf*)p;
  v8bf c1 = *(const v8bf*)(p + 8);
  return pack16(c0, c1);
}

// Butterfly reductions across the 16-lane half-wave that owns one matrix row.
static __device__ inline v8f half_bfly_max(v8f t) {
#pragma unroll
  for (int off = 1; off < 16; off <<= 1)
#pragma unroll
    for (int r = 0; r < 8; ++r) t[r] = fmaxf(t[r], __shfl_xor(t[r], off, 32));
  return t;
}
static __device__ inline v8f half_bfly_sum(v8f t) {
#pragma unroll
  for (int off = 1; off < 16; off <<= 1)
#pragma unroll
    for (int r = 0; r < 8; ++r) t[r] += __shfl_xor(t[r], off, 32);
  return t;
}

// ---------------------------------------------------------------------------
// Kernel 1a: W_{Q,K,V} [h][d][e] f32 -> WT [h][e][d] bf16 (column-major / head)
// ---------------------------------------------------------------------------
__global__ __launch_bounds__(256) void wt_hde_kernel(const float* __restrict__ W,
                                                     bf16_t* __restrict__ WT) {
  unsigned i = blockIdx.x * 256u + threadIdx.x;          // over 16*64*1024 = 1M
  if (i >= (unsigned)(N_HEADS * D_HEAD * D_MODEL)) return;
  int d = i & (D_MODEL - 1);
  int rest = i >> 10;
  int e = rest & (D_HEAD - 1);
  int h = rest >> 6;
  WT[i] = (__bf16)W[((size_t)h * D_MODEL + d) * D_HEAD + e];
}

// Kernel 1b: W_O [h][e][d] f32 (== row-major [k=he][n=d]) -> WTO [n=d][k] bf16
__global__ __launch_bounds__(256) void wt_o_kernel(const float* __restrict__ WO,
                                                   bf16_t* __restrict__ WT) {
  unsigned i = blockIdx.x * 256u + threadIdx.x;          // over 1024*1024
  if (i >= (unsigned)(D_MODEL * D_MODEL)) return;
  int k = i & (D_MODEL - 1);
  int d = i >> 10;
  WT[i] = (__bf16)WO[(size_t)k * D_MODEL + d];
}

// ---------------------------------------------------------------------------
// Kernel 2: QKV projection.  X [4096][1024] f32  @  WT [h][e][1024] bf16.
// One wave = 32(M) x 64(N=one head) tile; ping-pong pipelined K loop,
// 8 WMMAs per stage (16 per body).
// storeV==0 -> out [b][h][p][e] ; storeV==1 -> out [b][h][e][p] (V transposed)
// ---------------------------------------------------------------------------
__global__ __launch_bounds__(256) void qkv_proj_kernel(
    const float* __restrict__ X, const bf16_t* __restrict__ WT,
    const float* __restrict__ bias, bf16_t* __restrict__ out, int storeV) {
  const int lane = threadIdx.x & 31;
  const int wave = threadIdx.x >> 5;
  const int task = blockIdx.x * 8 + wave;     // 2048 tasks
  const int h = task >> 7;                    // head 0..15
  const int m0 = (task & 127) * 32;           // 32-row tile
  const int nl = lane & 15, hi = lane >> 4;

  const float* arow0 = X + (size_t)(m0 + nl) * D_MODEL;
  const float* arow1 = X + (size_t)(m0 + 16 + nl) * D_MODEL;
  const bf16_t* wh = WT + (size_t)h * D_HEAD * D_MODEL;

  v8f acc[2][4];
#pragma unroll
  for (int g = 0; g < 2; ++g)
#pragma unroll
    for (int c = 0; c < 4; ++c) acc[g][c] = vzero8();

  // prologue: buffer A <- kb = 0
  v16bf a0A = load_a_f32(arow0, 0, lane);
  v16bf a1A = load_a_f32(arow1, 0, lane);
  v16bf bA[4];
#pragma unroll
  for (int c = 0; c < 4; ++c)
    bA[c] = load_b_frag(wh + (size_t)(c * 16 + nl) * D_MODEL + (hi << 4));

  for (int kb = 0; kb < D_MODEL; kb += 64) {
    // stage 1: buffer B <- kb+32 while computing kb from buffer A
    v16bf a0B = load_a_f32(arow0, kb + 32, lane);
    v16bf a1B = load_a_f32(arow1, kb + 32, lane);
    v16bf bB[4];
#pragma unroll
    for (int c = 0; c < 4; ++c)
      bB[c] = load_b_frag(wh + (size_t)(c * 16 + nl) * D_MODEL + kb + 32 + (hi << 4));
#pragma unroll
    for (int c = 0; c < 4; ++c) acc[0][c] = wmma_bf16(a0A, bA[c], acc[0][c]);
#pragma unroll
    for (int c = 0; c < 4; ++c) acc[1][c] = wmma_bf16(a1A, bA[c], acc[1][c]);

    // stage 2: buffer A <- kb+64 (dummy reload of 0 on last) , compute kb+32
    const int kn = (kb + 64 < D_MODEL) ? kb + 64 : 0;
    a0A = load_a_f32(arow0, kn, lane);
    a1A = load_a_f32(arow1, kn, lane);
#pragma unroll
    for (int c = 0; c < 4; ++c)
      bA[c] = load_b_frag(wh + (size_t)(c * 16 + nl) * D_MODEL + kn + (hi << 4));
#pragma unroll
    for (int c = 0; c < 4; ++c) acc[0][c] = wmma_bf16(a0B, bB[c], acc[0][c]);
#pragma unroll
    for (int c = 0; c < 4; ++c) acc[1][c] = wmma_bf16(a1B, bB[c], acc[1][c]);
  }

  const int b = m0 >> 11;                 // batch
  const int p = m0 & (SEQ - 1);           // position
#pragma unroll
  for (int g = 0; g < 2; ++g) {
    const int prow = p + g * 16 + hi * 8;
#pragma unroll
    for (int c = 0; c < 4; ++c) {
      const int e = c * 16 + nl;
      const float bv = bias[h * D_HEAD + e];
      if (!storeV) {
        bf16_t* dst = out + (((size_t)(b * N_HEADS + h) * SEQ + prow) * D_HEAD) + e;
#pragma unroll
        for (int r = 0; r < 8; ++r) dst[(size_t)r * D_HEAD] = (__bf16)(acc[g][c][r] + bv);
      } else {
        bf16_t* dst = out + (((size_t)(b * N_HEADS + h) * D_HEAD + e) * SEQ) + prow;
        v8bf pk;
#pragma unroll
        for (int r = 0; r < 8; ++r) pk[r] = (__bf16)(acc[g][c][r] + bv);
        *(v8bf*)dst = pk;
      }
    }
  }
}

// ---------------------------------------------------------------------------
// Kernel 3: causal flash attention.  One wave = 16-query tile of one (b,h).
// Per 32-key block: K + V fragments loaded up front (V latency hidden behind
// softmax), 4 WMMAs (Q·Kt) -> scale/mask -> base-2 online softmax ->
// P transposed D->A layout through per-wave LDS -> 4 WMMAs (P·V).
// ---------------------------------------------------------------------------
__global__ __launch_bounds__(256) void attn_kernel(
    const bf16_t* __restrict__ Q, const bf16_t* __restrict__ K,
    const bf16_t* __restrict__ VT, bf16_t* __restrict__ Z) {
  __shared__ __align__(16) __bf16 plds[8][16 * 32];   // per-wave P staging

  const int lane = threadIdx.x & 31;
  const int wave = threadIdx.x >> 5;
  const int task = blockIdx.x * 8 + wave;       // 4096 = 32 (b,h) * 128 qtiles
  const int qtile = task & 127;
  const int bh = task >> 7;
  const int q0 = qtile * 16;
  const int nl = lane & 15, hi = lane >> 4;

  const bf16_t* qptr = Q + (size_t)bh * SEQ * D_HEAD;
  const bf16_t* kptr = K + (size_t)bh * SEQ * D_HEAD;
  const bf16_t* vptr = VT + (size_t)bh * D_HEAD * SEQ;

  // Q fragments held in registers for the whole key loop (d 0..31, 32..63).
  const bf16_t* qrow = qptr + (size_t)(q0 + nl) * D_HEAD;
  const v16bf qa0 = load_a_bf16(qrow, 0, lane);
  const v16bf qa1 = load_a_bf16(qrow, 32, lane);

  v8f acc[4];
#pragma unroll
  for (int c = 0; c < 4; ++c) acc[c] = vzero8();
  v8f mrow, lrow;
#pragma unroll
  for (int r = 0; r < 8; ++r) { mrow[r] = -3.0e38f; lrow[r] = 0.0f; }

  const int row_base = q0 + hi * 8;

  for (int jb = 0; jb < q0 + 16; jb += 32) {
    // ---- issue all global loads for this block up front
    const bf16_t* kp0 = kptr + (size_t)(jb + nl) * D_HEAD + (hi << 4);
    const bf16_t* kp1 = kptr + (size_t)(jb + 16 + nl) * D_HEAD + (hi << 4);
    v16bf kf0 = load_b_frag(kp0);
    v16bf kf1 = load_b_frag(kp0 + 32);
    v16bf kf2 = load_b_frag(kp1);
    v16bf kf3 = load_b_frag(kp1 + 32);
    v16bf vf[4];
#pragma unroll
    for (int c = 0; c < 4; ++c)
      vf[c] = load_b_frag(vptr + (size_t)(c * 16 + nl) * SEQ + jb + (hi << 4));

    // prefetch next key block (CDNA5 global_prefetch_b8)
    if (jb + 32 < q0 + 16) {
      __builtin_prefetch(kptr + (size_t)(jb + 32 + nl) * D_HEAD + (hi << 4), 0, 3);
      __builtin_prefetch(kptr + (size_t)(jb + 48 + nl) * D_HEAD + (hi << 4), 0, 3);
    }

    // ---- scores: two 16x16 tiles (keys jb..+15, jb+16..+31), K dim = d_head
    v8f s0 = vzero8(), s1 = vzero8();
    s0 = wmma_bf16(qa0, kf0, s0);
    s0 = wmma_bf16(qa1, kf1, s0);
    s1 = wmma_bf16(qa0, kf2, s1);
    s1 = wmma_bf16(qa1, kf3, s1);

    // ---- scale (log2 domain) + causal mask
#pragma unroll
    for (int r = 0; r < 8; ++r) {
      const int row = row_base + r;
      s0[r] = (jb + nl <= row) ? s0[r] * SCALE_LOG2 : NEG_INF;
      s1[r] = (jb + 16 + nl <= row) ? s1[r] * SCALE_LOG2 : NEG_INF;
    }

    // ---- base-2 online softmax
    v8f t;
#pragma unroll
    for (int r = 0; r < 8; ++r) t[r] = fmaxf(s0[r], s1[r]);
    t = half_bfly_max(t);
    v8f mnew, alpha, p0, p1;
#pragma unroll
    for (int r = 0; r < 8; ++r) {
      mnew[r] = fmaxf(mrow[r], t[r]);
      alpha[r] = __builtin_amdgcn_exp2f(mrow[r] - mnew[r]);
      p0[r] = __builtin_amdgcn_exp2f(s0[r] - mnew[r]);
      p1[r] = __builtin_amdgcn_exp2f(s1[r] - mnew[r]);
    }
    v8f rs;
#pragma unroll
    for (int r = 0; r < 8; ++r) rs[r] = p0[r] + p1[r];
    rs = half_bfly_sum(rs);
#pragma unroll
    for (int r = 0; r < 8; ++r) {
      lrow[r] = lrow[r] * alpha[r] + rs[r];
      mrow[r] = mnew[r];
    }
#pragma unroll
    for (int c = 0; c < 4; ++c)
#pragma unroll
      for (int r = 0; r < 8; ++r) acc[c][r] *= alpha[r];

    // ---- P: D-layout -> A-layout via per-wave LDS (no cross-wave sync needed)
    __bf16* pl = &plds[wave][0];
#pragma unroll
    for (int r = 0; r < 8; ++r) {
      const int row = r + hi * 8;
      pl[row * 32 + nl] = (__bf16)p0[r];
      pl[row * 32 + 16 + nl] = (__bf16)p1[r];
    }
    v16bf pa;
    {
      const __bf16* prow = &plds[wave][nl * 32];
      v8bf c0 = *(const v8bf*)(prow + hi * 16);
      v8bf c1 = *(const v8bf*)(prow + hi * 16 + 8);
      pa = pack16(c0, c1);
    }

    // ---- acc += P(16x32) @ V(32x64)  (vf loaded long ago; latency hidden)
#pragma unroll
    for (int c = 0; c < 4; ++c) acc[c] = wmma_bf16(pa, vf[c], acc[c]);
  }

  // ---- normalize, store z as [b][q][h*64+e] bf16
  const int b = bh >> 4, h = bh & 15;
  v8f inv;
#pragma unroll
  for (int r = 0; r < 8; ++r) inv[r] = 1.0f / lrow[r];
#pragma unroll
  for (int c = 0; c < 4; ++c) {
    const int col = h * D_HEAD + c * 16 + nl;
#pragma unroll
    for (int r = 0; r < 8; ++r) {
      const int row = row_base + r;
      Z[((size_t)b * SEQ + row) * D_MODEL + col] = (__bf16)(acc[c][r] * inv[r]);
    }
  }
}

// ---------------------------------------------------------------------------
// Kernel 4: output projection. Z [4096][1024] bf16 @ WTO [d][1024] bf16 + b_O.
// 32(M) x 64(N) per-wave tile; same ping-pong pipeline as kernel 2.
// ---------------------------------------------------------------------------
__global__ __launch_bounds__(256) void out_proj_kernel(
    const bf16_t* __restrict__ Z, const bf16_t* __restrict__ WTO,
    const float* __restrict__ bO, float* __restrict__ OUT) {
  const int lane = threadIdx.x & 31;
  const int wave = threadIdx.x >> 5;
  const int task = blockIdx.x * 8 + wave;     // 2048 tasks
  const int n0 = (task >> 7) * 64;
  const int m0 = (task & 127) * 32;
  const int nl = lane & 15, hi = lane >> 4;

  const bf16_t* arow0 = Z + (size_t)(m0 + nl) * D_MODEL;
  const bf16_t* arow1 = Z + (size_t)(m0 + 16 + nl) * D_MODEL;

  v8f acc[2][4];
#pragma unroll
  for (int g = 0; g < 2; ++g)
#pragma unroll
    for (int c = 0; c < 4; ++c) acc[g][c] = vzero8();

  v16bf a0A = load_a_bf16(arow0, 0, lane);
  v16bf a1A = load_a_bf16(arow1, 0, lane);
  v16bf bA[4];
#pragma unroll
  for (int c = 0; c < 4; ++c)
    bA[c] = load_b_frag(WTO + (size_t)(n0 + c * 16 + nl) * D_MODEL + (hi << 4));

  for (int kb = 0; kb < D_MODEL; kb += 64) {
    v16bf a0B = load_a_bf16(arow0, kb + 32, lane);
    v16bf a1B = load_a_bf16(arow1, kb + 32, lane);
    v16bf bB[4];
#pragma unroll
    for (int c = 0; c < 4; ++c)
      bB[c] = load_b_frag(WTO + (size_t)(n0 + c * 16 + nl) * D_MODEL + kb + 32 + (hi << 4));
#pragma unroll
    for (int c = 0; c < 4; ++c) acc[0][c] = wmma_bf16(a0A, bA[c], acc[0][c]);
#pragma unroll
    for (int c = 0; c < 4; ++c) acc[1][c] = wmma_bf16(a1A, bA[c], acc[1][c]);

    const int kn = (kb + 64 < D_MODEL) ? kb + 64 : 0;
    a0A = load_a_bf16(arow0, kn, lane);
    a1A = load_a_bf16(arow1, kn, lane);
#pragma unroll
    for (int c = 0; c < 4; ++c)
      bA[c] = load_b_frag(WTO + (size_t)(n0 + c * 16 + nl) * D_MODEL + kn + (hi << 4));
#pragma unroll
    for (int c = 0; c < 4; ++c) acc[0][c] = wmma_bf16(a0B, bB[c], acc[0][c]);
#pragma unroll
    for (int c = 0; c < 4; ++c) acc[1][c] = wmma_bf16(a1B, bB[c], acc[1][c]);
  }

#pragma unroll
  for (int g = 0; g < 2; ++g) {
#pragma unroll
    for (int c = 0; c < 4; ++c) {
      const int col = n0 + c * 16 + nl;
      const float bv = bO[col];
      float* dst = OUT + (size_t)(m0 + g * 16 + hi * 8) * D_MODEL + col;
#pragma unroll
      for (int r = 0; r < 8; ++r) dst[(size_t)r * D_MODEL] = acc[g][c][r] + bv;
    }
  }
}

// ---------------------------------------------------------------------------
// Host side
// ---------------------------------------------------------------------------
extern "C" void kernel_launch(void* const* d_in, const int* in_sizes, int n_in,
                              void* d_out, int out_size, void* d_ws, size_t ws_size,
                              hipStream_t stream) {
  (void)in_sizes; (void)n_in; (void)out_size; (void)ws_size;
  const float* Xq = (const float*)d_in[0];
  const float* Xk = (const float*)d_in[1];
  const float* Xv = (const float*)d_in[2];
  const float* WQ = (const float*)d_in[3];
  const float* WK = (const float*)d_in[4];
  const float* WV = (const float*)d_in[5];
  const float* WO = (const float*)d_in[6];
  const float* bQ = (const float*)d_in[7];
  const float* bK = (const float*)d_in[8];
  const float* bV = (const float*)d_in[9];
  const float* bO = (const float*)d_in[10];
  float* OUT = (float*)d_out;

  // Workspace layout (bf16), total 40 MB:
  char* ws = (char*)d_ws;
  const size_t WT_SZ = (size_t)N_HEADS * D_HEAD * D_MODEL * 2;     // 2 MB
  const size_t QK_SZ = (size_t)BATCH * N_HEADS * SEQ * D_HEAD * 2; // 8 MB
  bf16_t* wtq = (bf16_t*)(ws);
  bf16_t* wtk = (bf16_t*)(ws + WT_SZ);
  bf16_t* wtv = (bf16_t*)(ws + 2 * WT_SZ);
  bf16_t* wto = (bf16_t*)(ws + 3 * WT_SZ);
  bf16_t* qbuf = (bf16_t*)(ws + 4 * WT_SZ);
  bf16_t* kbuf = (bf16_t*)(ws + 4 * WT_SZ + QK_SZ);
  bf16_t* vtbuf = (bf16_t*)(ws + 4 * WT_SZ + 2 * QK_SZ);
  bf16_t* zbuf = (bf16_t*)(ws + 4 * WT_SZ + 3 * QK_SZ);

  // 1) weight prep (1M elems each -> 4096 blocks of 256)
  wt_hde_kernel<<<4096, 256, 0, stream>>>(WQ, wtq);
  wt_hde_kernel<<<4096, 256, 0, stream>>>(WK, wtk);
  wt_hde_kernel<<<4096, 256, 0, stream>>>(WV, wtv);
  wt_o_kernel<<<4096, 256, 0, stream>>>(WO, wto);

  // 2) QKV projections: 2048 wave-tasks = 256 blocks x 8 waves
  qkv_proj_kernel<<<256, 256, 0, stream>>>(Xq, wtq, bQ, qbuf, 0);
  qkv_proj_kernel<<<256, 256, 0, stream>>>(Xk, wtk, bK, kbuf, 0);
  qkv_proj_kernel<<<256, 256, 0, stream>>>(Xv, wtv, bV, vtbuf, 1);

  // 3) causal attention: 4096 wave-tasks = 512 blocks x 8 waves
  attn_kernel<<<512, 256, 0, stream>>>(qbuf, kbuf, vtbuf, zbuf);

  // 4) output projection: 2048 wave-tasks
  out_proj_kernel<<<256, 256, 0, stream>>>(zbuf, wto, bO, OUT);
}
